// PointNetSetAbstractionMsg_19353122636555
// MI455X (gfx1250) — compile-verified, matrix operand order checked by reading, not run on hardware
//
#include <hip/hip_runtime.h>
#include <hip/hip_bf16.h>

typedef __attribute__((ext_vector_type(16))) _Float16 v16h;
typedef __attribute__((ext_vector_type(8)))  _Float16 v8h;
typedef __attribute__((ext_vector_type(8)))  float    v8f;

#define NPTS   4096
#define NSAMP  512
#define NBATCH 8

// ---------------------------------------------------------------------------
// Farthest point sampling: one block per batch. Point cloud staged in LDS,
// per-thread min-dist in registers, block argmax per iteration (512 iters).
// Also writes new_xyz (B,3,S) directly into d_out.
// ---------------------------------------------------------------------------
__global__ __launch_bounds__(256) void kFPS(const float* __restrict__ xyz,
                                            int* __restrict__ fpsIdx,
                                            float* __restrict__ outNX) {
  __shared__ float sx[NPTS], sy[NPTS], sz[NPTS];
  __shared__ float rv[256];
  __shared__ int   ri[256];
  __shared__ int   sFar;
  const int b = blockIdx.x, t = threadIdx.x;
  const float* xb = xyz + (long)b * 3 * NPTS;
  for (int p = t; p < NPTS; p += 256) {
    sx[p] = xb[p]; sy[p] = xb[NPTS + p]; sz[p] = xb[2 * NPTS + p];
  }
  __syncthreads();
  float dist[16];
#pragma unroll
  for (int i = 0; i < 16; i++) dist[i] = 1e10f;
  int far = 0;
  for (int it = 0; it < NSAMP; it++) {
    if (t == 0) {
      fpsIdx[b * NSAMP + it] = far;
      outNX[(long)b * 3 * NSAMP + it]             = sx[far];
      outNX[(long)b * 3 * NSAMP + NSAMP + it]     = sy[far];
      outNX[(long)b * 3 * NSAMP + 2 * NSAMP + it] = sz[far];
    }
    const float cx = sx[far], cy = sy[far], cz = sz[far];
    float bv = -1.f; int bi = 0x7fffffff;
#pragma unroll
    for (int i = 0; i < 16; i++) {
      const int p = t + i * 256;
      const float dx = sx[p] - cx, dy = sy[p] - cy, dz = sz[p] - cz;
      const float d  = dx * dx + dy * dy + dz * dz;
      const float nd = fminf(dist[i], d);
      dist[i] = nd;
      if (nd > bv || (nd == bv && p < bi)) { bv = nd; bi = p; }
    }
    rv[t] = bv; ri[t] = bi; __syncthreads();
    for (int s2 = 128; s2 > 0; s2 >>= 1) {
      if (t < s2) {
        const float ov = rv[t + s2]; const int oi = ri[t + s2];
        if (ov > rv[t] || (ov == rv[t] && oi < ri[t])) { rv[t] = ov; ri[t] = oi; }
      }
      __syncthreads();
    }
    if (t == 0) sFar = ri[0];
    __syncthreads();
    far = sFar;
  }
}

// ---------------------------------------------------------------------------
// Ball query: one wave32 per centroid; ordered first-K selection via ballot
// prefix (== reference's sort-by-index then take first nsample, pad w/ first).
// ---------------------------------------------------------------------------
__global__ __launch_bounds__(256) void kBallQuery(const float* __restrict__ xyz,
                                                  const int* __restrict__ fpsIdx,
                                                  int* __restrict__ gidx,
                                                  float r2, int K) {
  const int g    = blockIdx.x * 8 + (threadIdx.x >> 5);
  const int lane = threadIdx.x & 31;
  const int b = g >> 9, s = g & (NSAMP - 1);
  const float* xb = xyz + (long)b * 3 * NPTS;
  const int ci = fpsIdx[b * NSAMP + s];
  const float cx = xb[ci], cy = xb[NPTS + ci], cz = xb[2 * NPTS + ci];
  int found = 0, first = -1;
  int* gout = gidx + (long)g * K;
  for (int base = 0; base < NPTS; base += 32) {
    const int p = base + lane;
    const float dx = xb[p] - cx, dy = xb[NPTS + p] - cy, dz = xb[2 * NPTS + p] - cz;
    const bool pred = (dx * dx + dy * dy + dz * dz) <= r2;
    const unsigned bal = (unsigned)__ballot(pred);
    if (first < 0 && bal) first = base + __builtin_ctz(bal);
    const int pos = found + __popc(bal & ((1u << lane) - 1u));
    if (pred && pos < K) gout[pos] = p;
    found += __popc(bal);
    if (found >= K) break;
  }
  if (first < 0) first = 0;
  for (int pos = (found < K ? found : K) + lane; pos < K; pos += 32) gout[pos] = first;
}

// ---------------------------------------------------------------------------
// Grouping: build X0 = [points[gidx], xyz[gidx]-new_xyz] as f16, C padded to 32.
// ---------------------------------------------------------------------------
__global__ void kGroup(const float* __restrict__ xyz, const float* __restrict__ pts,
                       const int* __restrict__ fpsIdx, const int* __restrict__ gidx,
                       _Float16* __restrict__ X0, int K) {
  const long row  = (long)blockIdx.x * blockDim.x + threadIdx.x;
  const long rows = (long)NBATCH * NSAMP * K;
  if (row >= rows) return;
  const long g = row / K;
  const int  s = (int)(g % NSAMP);
  const int  b = (int)(g / NSAMP);
  const int  p  = gidx[row];
  const int  ci = fpsIdx[b * NSAMP + s];
  const float* xb = xyz + (long)b * 3 * NPTS;
  const float* pb = pts + (long)b * 3 * NPTS;
  _Float16* o = X0 + row * 32;
  o[0] = (_Float16)pb[p];
  o[1] = (_Float16)pb[NPTS + p];
  o[2] = (_Float16)pb[2 * NPTS + p];
  o[3] = (_Float16)(xb[p] - xb[ci]);
  o[4] = (_Float16)(xb[NPTS + p] - xb[NPTS + ci]);
  o[5] = (_Float16)(xb[2 * NPTS + p] - xb[2 * NPTS + ci]);
#pragma unroll
  for (int c = 6; c < 32; c++) o[c] = (_Float16)0.f;
}

// ---------------------------------------------------------------------------
// Weight prep: f32 (O x C) -> f16 (O x Cpad), zero-padded K so WMMA chunks
// of 32 need no bounds logic.
// ---------------------------------------------------------------------------
__global__ void kPrepW(const float* __restrict__ w, _Float16* __restrict__ wp,
                       int O, int C, int Cpad) {
  const int i = blockIdx.x * blockDim.x + threadIdx.x;
  if (i < O * Cpad) {
    const int o = i / Cpad, c = i % Cpad;
    wp[i] = (c < C) ? (_Float16)w[o * C + c] : (_Float16)0.f;
  }
}

__global__ void kZero(float* p, int n) {
  const int i = blockIdx.x * blockDim.x + threadIdx.x;
  if (i < n) p[i] = 0.f;
}

__global__ void kIdent(float* sc, float* sh, int n) {
  const int i = threadIdx.x;
  if (i < n) { sc[i] = 1.f; sh[i] = 0.f; }
}

// BN fold: scale = gamma*rsqrt(var+eps); shift = beta - mean*scale.
__global__ void kFinalize(const float* __restrict__ sum, const float* __restrict__ sq,
                          const float* __restrict__ gamma, const float* __restrict__ beta,
                          float* __restrict__ sc, float* __restrict__ sh,
                          float invN, int O) {
  const int o = threadIdx.x;
  if (o < O) {
    const float m = sum[o] * invN;
    const float v = sq[o] * invN - m * m;
    const float r = rsqrtf(v + 1e-5f);
    const float a = gamma[o] * r;
    sc[o] = a;
    sh[o] = beta[o] - m * a;
  }
}

// ---------------------------------------------------------------------------
// WMMA GEMM: Y[m,o] = sum_c relu?(scIn[c]*X[m,c]+shIn[c]) * W[o,c] + bias[o].
// One wave computes a 16-row x O strip with v_wmma_f32_16x16x32_f16.
// Also accumulates per-channel sum / sumsq of the raw pre-activations
// (LDS reduce, then one global f32 atomic per channel per block) for BN.
// NT = O/16 output tiles, KC = Cpad/32 k-chunks, fully unrolled.
// ---------------------------------------------------------------------------
template <int NT, int KC>
__global__ __launch_bounds__(256) void kGemm(const _Float16* __restrict__ X,
                                             const _Float16* __restrict__ Wp,
                                             const float* __restrict__ bias,
                                             const float* __restrict__ scIn,
                                             const float* __restrict__ shIn,
                                             int reluIn,
                                             _Float16* __restrict__ Y,
                                             float* __restrict__ stSum,
                                             float* __restrict__ stSq) {
  constexpr int Cpad = KC * 32;
  constexpr int O    = NT * 16;
  __shared__ float sSc[Cpad], sSh[Cpad];
  __shared__ float sSum[O], sSq[O];
  const int t = threadIdx.x;
  if (t < Cpad) { sSc[t] = scIn[t]; sSh[t] = shIn[t]; }
  if (t < O)    { sSum[t] = 0.f;    sSq[t] = 0.f; }
  __syncthreads();

  const int wv = t >> 5, lane = t & 31;
  const int lm = lane & 15, lh = lane >> 4;
  const long m0 = ((long)blockIdx.x * 8 + wv) * 16;

  v8f acc[NT];
#pragma unroll
  for (int n = 0; n < NT; n++) {
    const float bo = bias[n * 16 + lm];
#pragma unroll
    for (int r = 0; r < 8; r++) acc[n][r] = bo;
  }

  const _Float16* xrow = X + (m0 + lm) * (long)Cpad;
#pragma unroll
  for (int kc = 0; kc < KC; kc++) {
    const int kb = kc * 32 + lh * 8;
    const v8h a0 = *(const v8h*)(xrow + kb);
    const v8h a1 = *(const v8h*)(xrow + kb + 16);
    v16h a;
#pragma unroll
    for (int j = 0; j < 8; j++) {
      float v0 = (float)a0[j] * sSc[kb + j]      + sSh[kb + j];
      float v1 = (float)a1[j] * sSc[kb + 16 + j] + sSh[kb + 16 + j];
      if (reluIn) { v0 = v0 > 0.f ? v0 : 0.f; v1 = v1 > 0.f ? v1 : 0.f; }
      a[j]     = (_Float16)v0;
      a[j + 8] = (_Float16)v1;
    }
#pragma unroll
    for (int n = 0; n < NT; n++) {
      const _Float16* wrow = Wp + (long)(n * 16 + lm) * Cpad + kb;
      const v8h b0 = *(const v8h*)(wrow);
      const v8h b1 = *(const v8h*)(wrow + 16);
      v16h bm;
#pragma unroll
      for (int j = 0; j < 8; j++) { bm[j] = b0[j]; bm[j + 8] = b1[j]; }
      acc[n] = __builtin_amdgcn_wmma_f32_16x16x32_f16(
          false, a, false, bm, (short)0, acc[n], false, false);
    }
  }

#pragma unroll
  for (int n = 0; n < NT; n++) {
    const int o = n * 16 + lm;
    float s = 0.f, q = 0.f;
#pragma unroll
    for (int r = 0; r < 8; r++) {
      const float v = acc[n][r];
      const long row = m0 + r + lh * 8;
      Y[row * O + o] = (_Float16)v;
      s += v; q += v * v;
    }
    s += __shfl_xor(s, 16, 32);
    q += __shfl_xor(q, 16, 32);
    if (lh == 0) { atomicAdd(&sSum[o], s); atomicAdd(&sSq[o], q); }
  }
  __syncthreads();
  if (t < O) { atomicAdd(&stSum[t], sSum[t]); atomicAdd(&stSq[t], sSq[t]); }
}

// ---------------------------------------------------------------------------
// Max-pool over K with BN affine + relu applied on the fly; writes (B,320,S).
// ---------------------------------------------------------------------------
__global__ void kMaxOut(const _Float16* __restrict__ X,
                        const float* __restrict__ sc, const float* __restrict__ sh,
                        float* __restrict__ out2, int K, int O, int chOff) {
  const long idx = (long)blockIdx.x * blockDim.x + threadIdx.x;
  const long tot = (long)NBATCH * NSAMP * O;
  if (idx >= tot) return;
  const int  o = (int)(idx % O);
  const long g = idx / O;
  const int  s = (int)(g % NSAMP);
  const int  b = (int)(g / NSAMP);
  const float a = sc[o], c = sh[o];
  const _Float16* xp = X + ((long)(b * NSAMP + s) * K) * O + o;
  float m = 0.f;  // relu output >= 0, so 0 is the identity for max(relu(.))
  for (int k = 0; k < K; k++) {
    float v = (float)xp[(long)k * O] * a + c;
    v = v > 0.f ? v : 0.f;
    m = v > m ? v : m;
  }
  out2[((long)b * 320 + chOff + o) * NSAMP + s] = m;
}

// ---------------------------------------------------------------------------
static inline void launch_gemm(int NT, int KC, int grid, hipStream_t st,
                               const _Float16* in, const _Float16* wpp,
                               const float* bias, const float* scIn,
                               const float* shIn, int relu, _Float16* outb,
                               float* stSum, float* stSq) {
#define GC(NTv, KCv)                                                          \
  if (NT == NTv && KC == KCv) {                                               \
    kGemm<NTv, KCv><<<grid, 256, 0, st>>>(in, wpp, bias, scIn, shIn, relu,    \
                                          outb, stSum, stSq);                 \
    return;                                                                   \
  }
  GC(2, 1) GC(4, 1) GC(4, 2) GC(6, 2) GC(8, 2) GC(8, 3)
#undef GC
}

extern "C" void kernel_launch(void* const* d_in, const int* in_sizes, int n_in,
                              void* d_out, int out_size, void* d_ws, size_t ws_size,
                              hipStream_t stream) {
  const float* xyz = (const float*)d_in[0];
  const float* pts = (const float*)d_in[1];

  static const int Cs[3][3] = {{6, 32, 32}, {6, 64, 64}, {6, 64, 96}};
  static const int Os[3][3] = {{32, 32, 64}, {64, 64, 128}, {64, 96, 128}};
  static const int KS[3]    = {16, 32, 128};
  static const float RAD[3] = {0.1f, 0.2f, 0.4f};
  static const int OFF[3]   = {0, 64, 192};

  // Parse param pointers; tolerate both pytree orders (sorted keys vs insertion).
  const float *Wv[3][3], *Bv[3][3], *Gm[3][3], *Bt[3][3];
  int pi = 2;
  for (int br = 0; br < 3; br++) {
    for (int ly = 0; ly < 3; ly++) {
      const int O = Os[br][ly], C = Cs[br][ly];
      if (pi + 3 < n_in && in_sizes[pi] == O * C) {  // w, b, gamma, beta
        Wv[br][ly] = (const float*)d_in[pi];
        Bv[br][ly] = (const float*)d_in[pi + 1];
        Gm[br][ly] = (const float*)d_in[pi + 2];
        Bt[br][ly] = (const float*)d_in[pi + 3];
      } else {  // b, beta, gamma, w (sorted dict keys)
        Bv[br][ly] = (const float*)d_in[pi];
        Bt[br][ly] = (const float*)d_in[pi + 1];
        Gm[br][ly] = (const float*)d_in[pi + 2];
        Wv[br][ly] = (const float*)d_in[pi + 3];
      }
      pi += 4;
    }
  }

  // Workspace carve-up.
  char* wsp = (char*)d_ws;
  auto alloc = [&](size_t bytes) -> char* {
    char* p = wsp;
    wsp += (bytes + 255) & ~(size_t)255;
    return p;
  };
  int*   fpsIdx = (int*)alloc((size_t)NBATCH * NSAMP * 4);
  int*   gidx   = (int*)alloc((size_t)NBATCH * NSAMP * 128 * 4);
  float* stSum  = (float*)alloc(128 * 4);
  float* stSq   = (float*)alloc(128 * 4);
  float* scA    = (float*)alloc(128 * 4);
  float* shA    = (float*)alloc(128 * 4);
  float* scB    = (float*)alloc(128 * 4);
  float* shB    = (float*)alloc(128 * 4);
  _Float16* wp[3][3];
  for (int br = 0; br < 3; br++)
    for (int ly = 0; ly < 3; ly++) {
      const int C = Cs[br][ly], O = Os[br][ly];
      const int Cp = ((C + 31) / 32) * 32;
      wp[br][ly] = (_Float16*)alloc((size_t)O * Cp * 2);
    }
  _Float16* bufA = (_Float16*)alloc((size_t)524288 * 96 * 2);   // X0 / X2
  _Float16* bufB = (_Float16*)alloc((size_t)524288 * 128 * 2);  // X1 / X3

  float* outNX = (float*)d_out;                    // (B,3,S)
  float* out2  = outNX + (long)NBATCH * 3 * NSAMP; // (B,320,S)

  // Stage 1: FPS (also emits new_xyz output).
  kFPS<<<NBATCH, 256, 0, stream>>>(xyz, fpsIdx, outNX);

  // Stage 2: weight conversion to padded f16.
  for (int br = 0; br < 3; br++)
    for (int ly = 0; ly < 3; ly++) {
      const int C = Cs[br][ly], O = Os[br][ly];
      const int Cp = ((C + 31) / 32) * 32;
      kPrepW<<<(O * Cp + 255) / 256, 256, 0, stream>>>(Wv[br][ly], wp[br][ly], O, C, Cp);
    }

  // Stage 3: per-branch pipeline.
  for (int br = 0; br < 3; br++) {
    const int K = KS[br];
    const long rows = (long)NBATCH * NSAMP * K;

    kBallQuery<<<(NBATCH * NSAMP) / 8, 256, 0, stream>>>(xyz, fpsIdx, gidx,
                                                         RAD[br] * RAD[br], K);
    kGroup<<<(int)((rows + 255) / 256), 256, 0, stream>>>(xyz, pts, fpsIdx, gidx,
                                                          bufA, K);
    kIdent<<<1, 128, 0, stream>>>(scA, shA, 128);

    float* scIn = scA;  float* shIn = shA;
    float* scOut = scB; float* shOut = shB;
    for (int ly = 0; ly < 3; ly++) {
      const int C = Cs[br][ly], O = Os[br][ly];
      const int Cp = ((C + 31) / 32) * 32;
      _Float16* in   = (ly % 2 == 0) ? bufA : bufB;
      _Float16* outb = (ly % 2 == 0) ? bufB : bufA;
      kZero<<<1, 128, 0, stream>>>(stSum, 128);
      kZero<<<1, 128, 0, stream>>>(stSq, 128);
      launch_gemm(O / 16, Cp / 32, (int)(rows / 128), stream, in, wp[br][ly],
                  Bv[br][ly], scIn, shIn, ly > 0 ? 1 : 0, outb, stSum, stSq);
      kFinalize<<<1, O, 0, stream>>>(stSum, stSq, Gm[br][ly], Bt[br][ly],
                                     scOut, shOut, 1.0f / (float)rows, O);
      float* ts = scIn; scIn = scOut; scOut = ts;
      float* th = shIn; shIn = shOut; shOut = th;
    }
    const int Olast = Os[br][2];
    kMaxOut<<<(NBATCH * NSAMP * Olast + 255) / 256, 256, 0, stream>>>(
        bufB, scIn, shIn, out2, K, Olast, OFF[br]);
  }
}
